// BiImprovedLSTM_49306224558690
// MI455X (gfx1250) — matrix operand order repaired and checked by reading, not compiled, not run on hardware
//
#include <hip/hip_runtime.h>

#define B_  32
#define T_  512
#define D_  256
#define H_  256
#define E_  512
#define G5H 1280   // 5*H
#define BT  (B_*T_)

typedef __bf16 bf16;
typedef __attribute__((ext_vector_type(16))) __bf16 v16bf;
typedef __attribute__((ext_vector_type(8)))  __bf16 bf16x8;
typedef __attribute__((ext_vector_type(8)))  float  v8f;

// ---------------------------------------------------------------- fragments
// A fragment: 16x32 (MxK) bf16, source row-major [m][k], row stride ld.
// ISA layout: lanes 0-15 -> M, VGPR0-3 K=0..7, VGPR4-7 K=16..23;
//             lanes 16-31 same M, K=8..15 / 24..31.
__device__ __forceinline__ v16bf load_fragA(const bf16* base, int ld) {
  const int lane = threadIdx.x & 31;
  const int m    = lane & 15;
  const int kb   = (lane & 16) ? 8 : 0;
  const bf16* p  = base + m * ld + kb;
  bf16x8 lo = *reinterpret_cast<const bf16x8*>(p);
  bf16x8 hi = *reinterpret_cast<const bf16x8*>(p + 16);
  v16bf f;
#pragma unroll
  for (int j = 0; j < 8; ++j) { f[j] = lo[j]; f[8 + j] = hi[j]; }
  return f;
}

// B fragment: 32x16 (KxN) bf16, source is B^T row-major [n][k], row stride ld.
// ISA layout: lanes 0-15 -> N with K=0..7 / 16..23, lanes 16-31 -> N with K=8..15 / 24..31.
__device__ __forceinline__ v16bf load_fragB(const bf16* base, int ld) {
  const int lane = threadIdx.x & 31;
  const int n    = lane & 15;
  const int kb   = (lane & 16) ? 8 : 0;
  const bf16* p  = base + n * ld + kb;
  bf16x8 lo = *reinterpret_cast<const bf16x8*>(p);
  bf16x8 hi = *reinterpret_cast<const bf16x8*>(p + 16);
  v16bf f;
#pragma unroll
  for (int j = 0; j < 8; ++j) { f[j] = lo[j]; f[8 + j] = hi[j]; }
  return f;
}

// C/D 16x16 f32: VGPR r -> row (r + 8*hi-half), lane%16 -> col.
__device__ __forceinline__ void store_tile(float* base, int ld, v8f c) {
  const int lane = threadIdx.x & 31;
  const int n    = lane & 15;
  const int r0   = (lane & 16) ? 8 : 0;
#pragma unroll
  for (int r = 0; r < 8; ++r) base[(size_t)(r0 + r) * ld + n] = c[r];
}

__device__ __forceinline__ v8f wmma_bf16(v16bf a, v16bf b, v8f c) {
  return __builtin_amdgcn_wmma_f32_16x16x32_bf16(false, a, false, b,
                                                 (short)0, c, false, false);
}

__device__ __forceinline__ float sigmoidf_(float x) {
  return 1.0f / (1.0f + __expf(-x));
}

// ---------------------------------------------------------------- prep
__global__ void k_f32_to_bf16(const float* __restrict__ s, bf16* __restrict__ d, int n) {
  int i = blockIdx.x * blockDim.x + threadIdx.x;
  if (i < n) d[i] = (bf16)s[i];
}

// Wt[((dir*5+g)*H + h)*D + i] = W_dir[(g*D + i)*H + h]   (N-major for B frags)
__global__ void k_prep_W(const float* __restrict__ Wf, const float* __restrict__ Wb,
                         bf16* __restrict__ Wt) {
  int idx = blockIdx.x * blockDim.x + threadIdx.x;
  if (idx >= 2 * 5 * H_ * D_) return;
  int i  = idx & (D_ - 1);
  int h  = (idx >> 8) & (H_ - 1);
  int dg = idx >> 16;               // dir*5+g  (H_*D_ == 65536)
  int dir = dg / 5, g = dg % 5;
  const float* W = dir ? Wb : Wf;
  Wt[idx] = (bf16)W[(g * D_ + i) * H_ + h];
}

// Ut[((dir*5+g)*H + o)*H + h] = U_dir[(g*H + h)*H + o]
__global__ void k_prep_U(const float* __restrict__ Uf, const float* __restrict__ Ub,
                         bf16* __restrict__ Ut) {
  int idx = blockIdx.x * blockDim.x + threadIdx.x;
  if (idx >= 2 * 5 * H_ * H_) return;
  int h  = idx & (H_ - 1);
  int o  = (idx >> 8) & (H_ - 1);
  int dg = idx >> 16;
  int dir = dg / 5, g = dg % 5;
  const float* U = dir ? Ub : Uf;
  Ut[idx] = (bf16)U[(g * H_ + h) * H_ + o];
}

// bias[(dir*5+g)*H + h] = bW + bU
__global__ void k_prep_bias(const float* __restrict__ bWf, const float* __restrict__ bUf,
                            const float* __restrict__ bWb, const float* __restrict__ bUb,
                            float* __restrict__ dst) {
  int idx = blockIdx.x * blockDim.x + threadIdx.x;
  if (idx >= 2 * 5 * H_) return;
  int gh = idx % (5 * H_);
  int dir = idx / (5 * H_);
  dst[idx] = dir ? (bWb[gh] + bUb[gh]) : (bWf[gh] + bUf[gh]);
}

// ---------------------------------------------------------------- xg = x*W + bW + bU
// 2x2 register tiling: one wave -> 32x32 output block, each fragment reused twice.
__global__ __launch_bounds__(256) void k_xg(const bf16* __restrict__ xb,
                                            const bf16* __restrict__ Wt,
                                            const float* __restrict__ bias,
                                            float* __restrict__ xg) {
  const int MT = BT / 32, NT = G5H / 32;              // 512 x 40 blocks per dir
  int wave = (blockIdx.x * blockDim.x + threadIdx.x) >> 5;
  if (wave >= 2 * MT * NT) return;
  const int dir = wave / (MT * NT); wave %= MT * NT;
  const int mt = wave / NT, nt = wave % NT;
  const int m0 = mt * 32, n0 = nt * 32;

  const bf16* ap = xb + (size_t)m0 * D_;
  const bf16* bp = Wt + ((size_t)dir * G5H + n0) * D_;
  v8f acc[2][2] = {{{}, {}}, {{}, {}}};
#pragma unroll
  for (int k = 0; k < D_ / 32; ++k) {
    if (k + 1 < D_ / 32) __builtin_prefetch(bp + (k + 1) * 32, 0, 1);
    v16bf a0 = load_fragA(ap + k * 32, D_);
    v16bf a1 = load_fragA(ap + (size_t)16 * D_ + k * 32, D_);
    v16bf b0 = load_fragB(bp + k * 32, D_);
    v16bf b1 = load_fragB(bp + (size_t)16 * D_ + k * 32, D_);
    acc[0][0] = wmma_bf16(a0, b0, acc[0][0]);
    acc[0][1] = wmma_bf16(a0, b1, acc[0][1]);
    acc[1][0] = wmma_bf16(a1, b0, acc[1][0]);
    acc[1][1] = wmma_bf16(a1, b1, acc[1][1]);
  }
  const int lane = threadIdx.x & 31;
#pragma unroll
  for (int j = 0; j < 2; ++j) {
    const float bv = bias[dir * G5H + n0 + j * 16 + (lane & 15)];
#pragma unroll
    for (int i = 0; i < 2; ++i) {
#pragma unroll
      for (int r = 0; r < 8; ++r) acc[i][j][r] += bv;
      store_tile(xg + (size_t)dir * BT * G5H + (size_t)(m0 + i * 16) * G5H + n0 + j * 16,
                 G5H, acc[i][j]);
    }
  }
}

// ---------------------------------------------------------------- recurrent LSTM
// grid = 4: (dir, batch-half).  16 waves, each owns (16 batches x 16 hidden) + 5 gates.
__global__ __launch_bounds__(512) void k_lstm(const float* __restrict__ xg,
                                              const bf16*  __restrict__ Ut,
                                              const float* __restrict__ dw,
                                              bf16* __restrict__ hcat) {
  const int dir  = blockIdx.x >> 1;
  const int b0   = (blockIdx.x & 1) * 16;
  const int wave = threadIdx.x >> 5;
  const int n0   = wave * 16;
  const int lane = threadIdx.x & 31;
  const int nn   = lane & 15;
  const int r0   = (lane & 16) ? 8 : 0;

  __shared__ __align__(16) bf16 hbuf[16][H_];
  for (int i = threadIdx.x; i < 16 * H_; i += blockDim.x)
    (&hbuf[0][0])[i] = (bf16)0.0f;
  v8f c = {};
  __syncthreads();

  const bf16* Ub = Ut + (size_t)dir * 5 * H_ * H_;

  for (int s = 0; s < T_; ++s) {
    const int t = dir ? (T_ - 1 - s) : s;

    v8f acc[5];
#pragma unroll
    for (int g = 0; g < 5; ++g) {
#pragma unroll
      for (int r = 0; r < 8; ++r) {
        int b = b0 + r0 + r;
        acc[g][r] = xg[((size_t)dir * BT + (size_t)b * T_ + t) * G5H + g * H_ + n0 + nn];
      }
    }
#pragma unroll
    for (int k2 = 0; k2 < H_ / 32; ++k2) {
      v16bf a = load_fragA(&hbuf[0][k2 * 32], H_);   // shared A across gates
#pragma unroll
      for (int g = 0; g < 5; ++g) {
        v16bf bb = load_fragB(Ub + ((size_t)g * H_ + n0) * H_ + k2 * 32, H_);
        acc[g] = wmma_bf16(a, bb, acc[g]);
      }
    }
    __syncthreads();                     // h reads done before overwrite

    float hl[8];
#pragma unroll
    for (int r = 0; r < 8; ++r) {
      int b = b0 + r0 + r;
      float gi = sigmoidf_(acc[0][r]);
      float gf = sigmoidf_(acc[1][r]);
      float go = sigmoidf_(acc[2][r]);
      float ch = tanhf(acc[3][r]);
      float gs = sigmoidf_(acc[4][r]) * dw[((size_t)b * T_ + t) * H_ + n0 + nn];
      float cn = gf * c[r] + gi * ch * gs;
      c[r] = cn;
      float hv = go * tanhf(cn);
      hl[r] = hv;
      hcat[((size_t)b * T_ + t) * E_ + dir * H_ + n0 + nn] = (bf16)hv;
    }
#pragma unroll
    for (int r = 0; r < 8; ++r) hbuf[r0 + r][n0 + nn] = (bf16)hl[r];
    __syncthreads();
  }
}

// ---------------------------------------------------------------- qkv projection
// 2x2 register tiling, scatter stores into q / k / v^T.
__global__ __launch_bounds__(256) void k_qkv(const bf16* __restrict__ hcat,
                                             const bf16* __restrict__ wq,
                                             const float* __restrict__ bq,
                                             bf16* __restrict__ q, bf16* __restrict__ kk,
                                             bf16* __restrict__ vT) {
  const int MT = BT / 32, NT = (3 * E_) / 32;   // 512 x 48
  int wave = (blockIdx.x * blockDim.x + threadIdx.x) >> 5;
  if (wave >= MT * NT) return;
  const int mt = wave / NT, nt = wave % NT;
  const int m0 = mt * 32, n0 = nt * 32;

  const bf16* ap = hcat + (size_t)m0 * E_;
  const bf16* bp = wq + (size_t)n0 * E_;
  v8f acc[2][2] = {{{}, {}}, {{}, {}}};
#pragma unroll
  for (int k = 0; k < E_ / 32; ++k) {
    v16bf a0 = load_fragA(ap + k * 32, E_);
    v16bf a1 = load_fragA(ap + (size_t)16 * E_ + k * 32, E_);
    v16bf b0 = load_fragB(bp + k * 32, E_);
    v16bf b1 = load_fragB(bp + (size_t)16 * E_ + k * 32, E_);
    acc[0][0] = wmma_bf16(a0, b0, acc[0][0]);
    acc[0][1] = wmma_bf16(a0, b1, acc[0][1]);
    acc[1][0] = wmma_bf16(a1, b0, acc[1][0]);
    acc[1][1] = wmma_bf16(a1, b1, acc[1][1]);
  }
  const int lane = threadIdx.x & 31;
  const int nn = lane & 15;
  const int r0 = (lane & 16) ? 8 : 0;
#pragma unroll
  for (int j = 0; j < 2; ++j) {
    const int n = n0 + j * 16 + nn;
    const float bv = bq[n];
#pragma unroll
    for (int i = 0; i < 2; ++i) {
#pragma unroll
      for (int r = 0; r < 8; ++r) {
        int m = m0 + i * 16 + r0 + r;
        int b_ = m >> 9, t = m & (T_ - 1);
        float v = acc[i][j][r] + bv;
        if (n < E_) {                                   // Q: [b][h][t][d]
          q[(((size_t)b_ * 4 + (n >> 7)) * T_ + t) * 128 + (n & 127)] = (bf16)v;
        } else if (n < 2 * E_) {                        // K: [b][h][t][d]
          int cix = n - E_;
          kk[(((size_t)b_ * 4 + (cix >> 7)) * T_ + t) * 128 + (cix & 127)] = (bf16)v;
        } else {                                        // V^T: [b][h][d][t]
          int cix = n - 2 * E_;
          vT[(((size_t)b_ * 4 + (cix >> 7)) * 128 + (cix & 127)) * T_ + t] = (bf16)v;
        }
      }
    }
  }
}

// ---------------------------------------------------------------- fused attention
// block = (b, head, 16 query rows); 8 waves.
__global__ __launch_bounds__(256) void k_attn(const bf16* __restrict__ q,
                                              const bf16* __restrict__ kk,
                                              const bf16* __restrict__ vT,
                                              bf16* __restrict__ aout) {
  const int idx = blockIdx.x;
  const int qt  = idx & 31;
  const int bh  = idx >> 5;                 // b*4 + head
  const int q0  = qt * 16;
  const int wave = threadIdx.x >> 5;

  __shared__ __align__(16) float Ssh[16][T_];
  __shared__ __align__(16) bf16  Psh[16][T_];
  __shared__ float red[16][16];
  __shared__ float rowm[16], rows[16];

  const bf16* qbase = q  + ((size_t)bh * T_ + q0) * 128;
  const bf16* kbase = kk + (size_t)bh * T_ * 128;

  // phase 1: S = q k^T / sqrt(d).  Q fragments hoisted and reused over k-tiles.
  {
    v16bf aq[4];
#pragma unroll
    for (int k2 = 0; k2 < 4; ++k2) aq[k2] = load_fragA(qbase + k2 * 32, 128);
    for (int nt = wave; nt < T_ / 16; nt += 8) {
      v8f acc = {};
#pragma unroll
      for (int k2 = 0; k2 < 4; ++k2) {
        v16bf b = load_fragB(kbase + (size_t)nt * 16 * 128 + k2 * 32, 128);
        acc = wmma_bf16(aq[k2], b, acc);
      }
      const float sc = 0.08838834764831845f;   // 1/sqrt(128)
#pragma unroll
      for (int r = 0; r < 8; ++r) acc[r] *= sc;
      store_tile(&Ssh[0][nt * 16], T_, acc);
    }
  }
  __syncthreads();

  // phase 2: row softmax (16 lanes per row)
  const int row = threadIdx.x >> 4;
  const int l16 = threadIdx.x & 15;
  float mx = -3.4e38f;
  for (int j = l16; j < T_; j += 16) mx = fmaxf(mx, Ssh[row][j]);
  red[row][l16] = mx;
  __syncthreads();
  if (l16 == 0) {
    float m = red[row][0];
    for (int j = 1; j < 16; ++j) m = fmaxf(m, red[row][j]);
    rowm[row] = m;
  }
  __syncthreads();
  const float mrow = rowm[row];
  float sum = 0.0f;
  for (int j = l16; j < T_; j += 16) {
    float e = __expf(Ssh[row][j] - mrow);
    Ssh[row][j] = e;
    sum += e;
  }
  red[row][l16] = sum;
  __syncthreads();
  if (l16 == 0) {
    float s = 0.0f;
    for (int j = 0; j < 16; ++j) s += red[row][j];
    rows[row] = 1.0f / s;
  }
  __syncthreads();
  const float inv = rows[row];
  for (int j = l16; j < T_; j += 16) Psh[row][j] = (bf16)(Ssh[row][j] * inv);
  __syncthreads();

  // phase 3: O = P v  (wave -> 16 of 128 head-dim cols)
  {
    const int nt = wave;
    v8f acc = {};
    const bf16* vb = vT + ((size_t)bh * 128 + nt * 16) * T_;
#pragma unroll
    for (int k2 = 0; k2 < T_ / 32; ++k2) {
      v16bf a = load_fragA(&Psh[0][k2 * 32], T_);
      v16bf b = load_fragB(vb + k2 * 32, T_);
      acc = wmma_bf16(a, b, acc);
    }
    const int b_ = bh >> 2, h_ = bh & 3;
    const int lane = threadIdx.x & 31;
    const int n = lane & 15;
    const int r0 = (lane & 16) ? 8 : 0;
#pragma unroll
    for (int r = 0; r < 8; ++r)
      aout[((size_t)b_ * T_ + q0 + r0 + r) * E_ + h_ * 128 + nt * 16 + n] = (bf16)acc[r];
  }
}

// ---------------------------------------------------------------- output projection
// 2x2 register tiling.
__global__ __launch_bounds__(256) void k_outproj(const bf16* __restrict__ aout,
                                                 const bf16* __restrict__ wo,
                                                 const float* __restrict__ bo,
                                                 float* __restrict__ out) {
  const int MT = BT / 32, NT = E_ / 32;           // 512 x 16
  int wave = (blockIdx.x * blockDim.x + threadIdx.x) >> 5;
  if (wave >= MT * NT) return;
  const int mt = wave / NT, nt = wave % NT;
  const int m0 = mt * 32, n0 = nt * 32;

  const bf16* ap = aout + (size_t)m0 * E_;
  const bf16* bp = wo + (size_t)n0 * E_;
  v8f acc[2][2] = {{{}, {}}, {{}, {}}};
#pragma unroll
  for (int k = 0; k < E_ / 32; ++k) {
    v16bf a0 = load_fragA(ap + k * 32, E_);
    v16bf a1 = load_fragA(ap + (size_t)16 * E_ + k * 32, E_);
    v16bf b0 = load_fragB(bp + k * 32, E_);
    v16bf b1 = load_fragB(bp + (size_t)16 * E_ + k * 32, E_);
    acc[0][0] = wmma_bf16(a0, b0, acc[0][0]);
    acc[0][1] = wmma_bf16(a0, b1, acc[0][1]);
    acc[1][0] = wmma_bf16(a1, b0, acc[1][0]);
    acc[1][1] = wmma_bf16(a1, b1, acc[1][1]);
  }
  const int lane = threadIdx.x & 31;
#pragma unroll
  for (int j = 0; j < 2; ++j) {
    const float bv = bo[n0 + j * 16 + (lane & 15)];
#pragma unroll
    for (int i = 0; i < 2; ++i) {
#pragma unroll
      for (int r = 0; r < 8; ++r) acc[i][j][r] += bv;
      store_tile(out + (size_t)(m0 + i * 16) * E_ + n0 + j * 16, E_, acc[i][j]);
    }
  }
}

// ---------------------------------------------------------------- layernorm (in place)
__global__ __launch_bounds__(256) void k_ln(float* __restrict__ o,
                                            const float* __restrict__ g,
                                            const float* __restrict__ bta) {
  float* p = o + (size_t)blockIdx.x * E_;
  __shared__ float r1[256], r2[256];
  float a = p[threadIdx.x], c = p[threadIdx.x + 256];
  r1[threadIdx.x] = a + c;
  r2[threadIdx.x] = a * a + c * c;
  __syncthreads();
  for (int st = 128; st > 0; st >>= 1) {
    if ((int)threadIdx.x < st) {
      r1[threadIdx.x] += r1[threadIdx.x + st];
      r2[threadIdx.x] += r2[threadIdx.x + st];
    }
    __syncthreads();
  }
  const float mu  = r1[0] * (1.0f / E_);
  const float var = r2[0] * (1.0f / E_) - mu * mu;
  const float inv = rsqrtf(var + 1e-5f);
  p[threadIdx.x]       = (a - mu) * inv * g[threadIdx.x]       + bta[threadIdx.x];
  p[threadIdx.x + 256] = (c - mu) * inv * g[threadIdx.x + 256] + bta[threadIdx.x + 256];
}

// ---------------------------------------------------------------- launch
extern "C" void kernel_launch(void* const* d_in, const int* in_sizes, int n_in,
                              void* d_out, int out_size, void* d_ws, size_t ws_size,
                              hipStream_t stream) {
  const float* x        = (const float*)d_in[0];
  const float* gw       = (const float*)d_in[1];
  const float* W_fwd    = (const float*)d_in[2];
  const float* bW_fwd   = (const float*)d_in[3];
  const float* U_fwd    = (const float*)d_in[4];
  const float* bU_fwd   = (const float*)d_in[5];
  const float* W_bwd    = (const float*)d_in[6];
  const float* bW_bwd   = (const float*)d_in[7];
  const float* U_bwd    = (const float*)d_in[8];
  const float* bU_bwd   = (const float*)d_in[9];
  const float* in_w     = (const float*)d_in[10];
  const float* in_b     = (const float*)d_in[11];
  const float* out_w    = (const float*)d_in[12];
  const float* out_b    = (const float*)d_in[13];
  const float* ln_g     = (const float*)d_in[14];
  const float* ln_b     = (const float*)d_in[15];
  float* out = (float*)d_out;

  char* ws = (char*)d_ws;
  size_t off = 0;
  auto alloc = [&](size_t bytes) -> char* {
    char* p = ws + off;
    off = (off + bytes + 255) & ~(size_t)255;
    return p;
  };
  bf16*  xb   = (bf16*)alloc((size_t)BT * D_ * 2);
  bf16*  Wt   = (bf16*)alloc((size_t)2 * 5 * H_ * D_ * 2);
  bf16*  Ut   = (bf16*)alloc((size_t)2 * 5 * H_ * H_ * 2);
  float* bias = (float*)alloc((size_t)2 * G5H * 4);
  bf16*  inw  = (bf16*)alloc((size_t)3 * E_ * E_ * 2);
  bf16*  outw = (bf16*)alloc((size_t)E_ * E_ * 2);
  float* xg   = (float*)alloc((size_t)2 * BT * G5H * 4);
  bf16*  hcat = (bf16*)alloc((size_t)BT * E_ * 2);
  bf16*  qb   = (bf16*)alloc((size_t)B_ * 4 * T_ * 128 * 2);
  bf16*  kb   = (bf16*)alloc((size_t)B_ * 4 * T_ * 128 * 2);
  bf16*  vT   = (bf16*)alloc((size_t)B_ * 4 * 128 * T_ * 2);
  bf16*  aout = (bf16*)alloc((size_t)BT * E_ * 2);

  int n;
  n = BT * D_;
  k_f32_to_bf16<<<(n + 255) / 256, 256, 0, stream>>>(x, xb, n);
  n = 2 * 5 * H_ * D_;
  k_prep_W<<<(n + 255) / 256, 256, 0, stream>>>(W_fwd, W_bwd, Wt);
  n = 2 * 5 * H_ * H_;
  k_prep_U<<<(n + 255) / 256, 256, 0, stream>>>(U_fwd, U_bwd, Ut);
  n = 2 * G5H;
  k_prep_bias<<<(n + 255) / 256, 256, 0, stream>>>(bW_fwd, bU_fwd, bW_bwd, bU_bwd, bias);
  n = 3 * E_ * E_;
  k_f32_to_bf16<<<(n + 255) / 256, 256, 0, stream>>>(in_w, inw, n);
  n = E_ * E_;
  k_f32_to_bf16<<<(n + 255) / 256, 256, 0, stream>>>(out_w, outw, n);

  // x*W + biases : 2 * 512 * 40 wave-blocks / 8 waves per block
  k_xg<<<(2 * (BT / 32) * (G5H / 32)) / 8, 256, 0, stream>>>(xb, Wt, bias, xg);

  // recurrent LSTM: (2 dirs) x (2 batch halves), 16 waves each
  k_lstm<<<4, 512, 0, stream>>>(xg, Ut, gw, hcat);

  // qkv projection: 512 * 48 wave-blocks / 8
  k_qkv<<<((BT / 32) * ((3 * E_) / 32)) / 8, 256, 0, stream>>>(hcat, inw, in_b, qb, kb, vT);

  // fused attention: 32 b * 4 heads * 32 q-tiles
  k_attn<<<B_ * 4 * (T_ / 16), 256, 0, stream>>>(qb, kb, vT, aout);

  // output projection: 512 * 16 wave-blocks / 8
  k_outproj<<<((BT / 32) * (E_ / 32)) / 8, 256, 0, stream>>>(aout, outw, out_b, out);

  // layernorm in place
  k_ln<<<BT, 256, 0, stream>>>(out, ln_g, ln_b);
}
